// MultiHeadGraphAttention_2989297238394
// MI455X (gfx1250) — matrix operand order repaired and checked by reading, not compiled, not run on hardware
//
#include <hip/hip_runtime.h>

#define B_SZ 4
#define N_SZ 2048
#define M_SZ 2048
#define D_SZ 512
#define H_SZ 8
#define HD_SZ 64
#define LN_EPS 1e-5f

typedef __attribute__((ext_vector_type(16))) __bf16 v16bf;
typedef __attribute__((ext_vector_type(8)))  float  v8f;

union frag_u { v16bf bf; uint4 q[2]; };

// ---------- bf16 helpers ----------
static __device__ __forceinline__ unsigned short f2bf_bits_rne(float f) {
  unsigned u = __builtin_bit_cast(unsigned, f);
  unsigned r = u + 0x7FFFu + ((u >> 16) & 1u);
  return (unsigned short)(r >> 16);
}
static __device__ __forceinline__ unsigned short f2bf_bits(float f) {
  __bf16 h = (__bf16)f;   // hardware cvt on gfx1250
  return __builtin_bit_cast(unsigned short, h);
}

static __device__ __forceinline__ v8f wmma_bf16(v16bf a, v16bf b, v8f c) {
  return __builtin_amdgcn_wmma_f32_16x16x32_bf16(
      /*neg_a=*/false, a, /*neg_b=*/false, b,
      /*c_mod=*/(short)0, c, /*reuse_a=*/false, /*reuse_b=*/false);
}

// ---------- async global -> LDS copy (CDNA5; tracked by ASYNCcnt) ----------
static __device__ __forceinline__ void async_copy_b128(unsigned lds_byte_addr,
                                                       unsigned long long gaddr) {
  asm volatile("global_load_async_to_lds_b128 %0, %1, off"
               :
               : "v"(lds_byte_addr), "v"(gaddr)
               : "memory");
}
static __device__ __forceinline__ void wait_asynccnt0() {
  asm volatile("s_wait_asynccnt 0" ::: "memory");
}
static __device__ __forceinline__ unsigned lds_addr32(const void* p) {
  return (unsigned)(size_t)p;  // shared aperture keeps LDS offset in addr[31:0]
}

// ---------- fragment loaders (ISA 7.12.2 layouts, wave32) ----------
// A 16x32 (MxK), row-major bf16 source, k contiguous. lane: m=L&15, h=L>>4.
// lane holds k = 8h+[0..7] (regs 0-3) and 16+8h+[0..7] (regs 4-7): two b128 loads.
static __device__ __forceinline__ v16bf load_a_bf16(const unsigned short* p, int ld) {
  int lane = threadIdx.x & 31;
  int m = lane & 15, hh = lane >> 4;
  const unsigned short* row = p + (size_t)m * ld + 8 * hh;
  frag_u u;
  u.q[0] = *(const uint4*)(row);
  u.q[1] = *(const uint4*)(row + 16);
  return u.bf;
}
// B 32x16 (KxN) with B[k][n] = src[n][k] (row-major [n][k], k contiguous).
// lane n holds k = 8h+[0..7] and 16+8h+[0..7]: two b128 loads.
static __device__ __forceinline__ v16bf load_bT_bf16(const unsigned short* p, int ld) {
  int lane = threadIdx.x & 31;
  int n = lane & 15, hh = lane >> 4;
  const unsigned short* row = p + (size_t)n * ld + 8 * hh;
  frag_u u;
  u.q[0] = *(const uint4*)(row);
  u.q[1] = *(const uint4*)(row + 16);
  return u.bf;
}

// ---------- kernel 0: fp32 -> bf16 bulk convert (runs once, off hot path) ----------
__global__ __launch_bounds__(256) void f32_to_bf16_kernel(
    const float* __restrict__ src, unsigned short* __restrict__ dst, int nelem) {
  int i = (blockIdx.x * blockDim.x + threadIdx.x) * 2;
  if (i < nelem) {
    float2 v = *(const float2*)(src + i);
    unsigned lo = f2bf_bits_rne(v.x);
    unsigned hi = f2bf_bits_rne(v.y);
    *(unsigned*)(dst + i) = lo | (hi << 16);
  }
}

// ---------- kernel 1: fused Q/K/V projection (bf16 in/out) ----------
// Q: scale folded in, layout [B,H,S,HD]. K: [B,H,S,HD]. V: transposed [B,H,HD,M].
__global__ __launch_bounds__(256) void qkv_proj_kernel(
    const unsigned short* __restrict__ qbf, const unsigned short* __restrict__ kbf,
    const unsigned short* __restrict__ vbf,
    const unsigned short* __restrict__ WQb, const unsigned short* __restrict__ WKb,
    const unsigned short* __restrict__ WVb,
    unsigned short* __restrict__ Qw, unsigned short* __restrict__ Kw,
    unsigned short* __restrict__ Vt) {
  int which = blockIdx.y;
  const unsigned short* X = (which == 0) ? qbf : (which == 1) ? kbf : vbf;
  const unsigned short* W = (which == 0) ? WQb : (which == 1) ? WKb : WVb;

  int wave = threadIdx.x >> 5;
  int lane = threadIdx.x & 31;
  int n = lane & 15, hh = lane >> 4;
  int rowbase = blockIdx.x * 16;   // flat row over [B*N)
  int colbase = wave * 64;         // 8 waves x 64 cols = 512

  v8f acc[4] = {};
  for (int k0 = 0; k0 < D_SZ; k0 += 32) {
    v16bf a = load_a_bf16(X + (size_t)rowbase * D_SZ + k0, D_SZ);
#pragma unroll
    for (int t = 0; t < 4; ++t) {
      v16bf b = load_bT_bf16(W + (size_t)(colbase + t * 16) * D_SZ + k0, D_SZ);
      acc[t] = wmma_bf16(a, b, acc[t]);
    }
  }
  float sc = (which == 0) ? 0.125f : 1.0f;  // 1/sqrt(HD) folded into Q
#pragma unroll
  for (int t = 0; t < 4; ++t) {
#pragma unroll
    for (int r = 0; r < 8; ++r) {
      int o = colbase + t * 16 + n;
      int head = o >> 6, hd = o & 63;
      int rg = rowbase + r + 8 * hh;
      int bb = rg >> 11, s = rg & 2047;
      unsigned short bits = f2bf_bits(acc[t][r] * sc);
      if (which == 2)  // V transposed: [B,H,HD,M]
        Vt[(((size_t)bb * H_SZ + head) * HD_SZ + hd) * M_SZ + s] = bits;
      else if (which == 1)
        Kw[(((size_t)bb * H_SZ + head) * N_SZ + s) * HD_SZ + hd] = bits;
      else
        Qw[(((size_t)bb * H_SZ + head) * N_SZ + s) * HD_SZ + hd] = bits;
    }
  }
}

// ---------- kernel 2: flash attention with async LDS staging of K/V ----------
__global__ __launch_bounds__(128) void attn_kernel(
    const unsigned short* __restrict__ Qw, const unsigned short* __restrict__ Kw,
    const unsigned short* __restrict__ Vt, const int* __restrict__ mask,
    unsigned short* __restrict__ AO) {
  __shared__ __align__(16) unsigned short ldsK[32 * 64];   // [key][hd], 4 KB
  __shared__ __align__(16) unsigned short ldsV[64 * 32];   // [hd][key], 4 KB
  __shared__ __align__(16) unsigned short ldsP[4][16][32]; // per-wave P, 4 KB
  int tid = threadIdx.x;
  int wave = tid >> 5;
  int lane = tid & 31;
  int n = lane & 15, hh = lane >> 4;
  int b = blockIdx.z;
  int h = blockIdx.y;
  int qbase = blockIdx.x * 64 + wave * 16;

  const unsigned short* Qp =
      Qw + (((size_t)b * H_SZ + h) * N_SZ + qbase) * HD_SZ;
  v16bf qf0 = load_a_bf16(Qp, HD_SZ);        // K = 0..31 of HD
  v16bf qf1 = load_a_bf16(Qp + 32, HD_SZ);   // K = 32..63

  const unsigned short* Kbase = Kw + ((size_t)b * H_SZ + h) * M_SZ * HD_SZ;
  const unsigned short* Vbase = Vt + ((size_t)b * H_SZ + h) * HD_SZ * M_SZ;
  unsigned dstK = lds_addr32(&ldsK[0]);
  unsigned dstV = lds_addr32(&ldsV[0]);

  v8f oacc[4] = {};
  float rm[8], rs[8];
#pragma unroll
  for (int r = 0; r < 8; ++r) { rm[r] = -1e30f; rs[r] = 0.0f; }

  for (int m0 = 0; m0 < M_SZ; m0 += 32) {
    // ---- stage K (contiguous 4 KB) and V (64 rows x 64 B) into LDS ----
    {
      unsigned long long ksrc =
          (unsigned long long)(size_t)(Kbase + (size_t)m0 * HD_SZ);
#pragma unroll
      for (int c = 0; c < 2; ++c) {
        int chunk = tid + c * 128;                       // 256 x 16 B = 4 KB
        async_copy_b128(dstK + chunk * 16, ksrc + (unsigned)(chunk * 16));
      }
      unsigned long long vsrc = (unsigned long long)(size_t)(Vbase + m0);
#pragma unroll
      for (int c = 0; c < 2; ++c) {
        int chunk = tid + c * 128;                       // 64 rows x 4 chunks
        int row = chunk >> 2, off = (chunk & 3) * 16;    // bytes within row
        async_copy_b128(dstV + row * 64 + off,
                        vsrc + (unsigned long long)row * (M_SZ * 2) + off);
      }
      wait_asynccnt0();
    }
    __syncthreads();

    // ---- S = Q K^T from LDS ----
    v8f s[2];
#pragma unroll
    for (int t = 0; t < 2; ++t) {
      const unsigned short* Kp = &ldsK[(size_t)(16 * t) * HD_SZ];
      v8f st = {};
      st = wmma_bf16(qf0, load_bT_bf16(Kp, HD_SZ), st);
      st = wmma_bf16(qf1, load_bT_bf16(Kp + 32, HD_SZ), st);
#pragma unroll
      for (int r = 0; r < 8; ++r) {
        int row = qbase + r + 8 * hh;
        int mv = mask[((size_t)b * N_SZ + row) * M_SZ + (m0 + 16 * t + n)];
        st[r] = mv ? st[r] : -1e30f;   // scale already folded into Q
      }
      s[t] = st;
    }
    // ---- online softmax: row stats live in 16-lane half-wave groups ----
    float tmax[8], psum[8], resc[8];
#pragma unroll
    for (int r = 0; r < 8; ++r) tmax[r] = fmaxf(s[0][r], s[1][r]);
#pragma unroll
    for (int xm = 1; xm < 16; xm <<= 1)
#pragma unroll
      for (int r = 0; r < 8; ++r)
        tmax[r] = fmaxf(tmax[r], __shfl_xor(tmax[r], xm, 32));
#pragma unroll
    for (int r = 0; r < 8; ++r) {
      float nm = fmaxf(rm[r], tmax[r]);
      resc[r] = __expf(rm[r] - nm);
      rm[r] = nm;
      psum[r] = 0.0f;
    }
#pragma unroll
    for (int t = 0; t < 2; ++t)
#pragma unroll
      for (int r = 0; r < 8; ++r) {
        float p = __expf(s[t][r] - rm[r]);
        s[t][r] = p;
        psum[r] += p;
      }
#pragma unroll
    for (int xm = 1; xm < 16; xm <<= 1)
#pragma unroll
      for (int r = 0; r < 8; ++r) psum[r] += __shfl_xor(psum[r], xm, 32);
#pragma unroll
    for (int r = 0; r < 8; ++r) rs[r] = rs[r] * resc[r] + psum[r];
#pragma unroll
    for (int t = 0; t < 4; ++t)
#pragma unroll
      for (int r = 0; r < 8; ++r) oacc[t][r] *= resc[r];

    // ---- C-layout P -> LDS -> A-layout (per-wave buffer; DS in-order) ----
#pragma unroll
    for (int t = 0; t < 2; ++t)
#pragma unroll
      for (int r = 0; r < 8; ++r)
        ldsP[wave][r + 8 * hh][16 * t + n] = f2bf_bits(s[t][r]);
    v16bf pa = load_a_bf16(&ldsP[wave][0][0], 32);

    // ---- O += P V from LDS V tile ([hd][key], k-contiguous for B-frag) ----
#pragma unroll
    for (int t = 0; t < 4; ++t) {
      v16bf bv = load_bT_bf16(&ldsV[(size_t)(t * 16) * 32], 32);
      oacc[t] = wmma_bf16(pa, bv, oacc[t]);
    }
    __syncthreads();  // protect ldsK/ldsV before next iteration's staging
  }
  // ---- normalize and store bf16 [B, N, 512] ----
#pragma unroll
  for (int t = 0; t < 4; ++t)
#pragma unroll
    for (int r = 0; r < 8; ++r) {
      float o = oacc[t][r] / rs[r];
      int row = qbase + r + 8 * hh;
      int col = h * HD_SZ + t * 16 + n;
      AO[((size_t)b * N_SZ + row) * D_SZ + col] = f2bf_bits(o);
    }
}

// ---------- kernel 3: output projection + bias + residual + LayerNorm ----------
__global__ __launch_bounds__(256) void oproj_ln_kernel(
    const unsigned short* __restrict__ AO, const unsigned short* __restrict__ WOb,
    const float* __restrict__ bO, const float* __restrict__ query,
    const float* __restrict__ gamma, const float* __restrict__ beta,
    float* __restrict__ out) {
  __shared__ float xbuf[16][D_SZ];
  int wave = threadIdx.x >> 5;
  int lane = threadIdx.x & 31;
  int n = lane & 15, hh = lane >> 4;
  int rowbase = blockIdx.x * 16;  // flat over [B*N)
  int colbase = wave * 64;

  v8f acc[4] = {};
  for (int k0 = 0; k0 < D_SZ; k0 += 32) {
    v16bf a = load_a_bf16(AO + (size_t)rowbase * D_SZ + k0, D_SZ);
#pragma unroll
    for (int t = 0; t < 4; ++t) {
      v16bf bw = load_bT_bf16(WOb + (size_t)(colbase + t * 16) * D_SZ + k0, D_SZ);
      acc[t] = wmma_bf16(a, bw, acc[t]);
    }
  }
#pragma unroll
  for (int t = 0; t < 4; ++t)
#pragma unroll
    for (int r = 0; r < 8; ++r) {
      int col = colbase + t * 16 + n;
      int row = r + 8 * hh;
      int rg = rowbase + row;
      float x = acc[t][r] + bO[col] + query[(size_t)rg * D_SZ + col];
      xbuf[row][col] = x;
    }
  __syncthreads();

  int row = threadIdx.x >> 4;   // 16 rows, 16 lanes per row
  int lid = threadIdx.x & 15;
  float s = 0.0f, s2 = 0.0f;
  for (int c = lid; c < D_SZ; c += 16) {
    float v = xbuf[row][c];
    s += v;
    s2 += v * v;
  }
#pragma unroll
  for (int xm = 1; xm < 16; xm <<= 1) {
    s  += __shfl_xor(s, xm, 32);
    s2 += __shfl_xor(s2, xm, 32);
  }
  float mean = s * (1.0f / D_SZ);
  float var = s2 * (1.0f / D_SZ) - mean * mean;
  float inv = rsqrtf(var + LN_EPS);
  int rg = rowbase + row;
  for (int c = lid; c < D_SZ; c += 16)
    out[(size_t)rg * D_SZ + c] = (xbuf[row][c] - mean) * inv * gamma[c] + beta[c];
}

extern "C" void kernel_launch(void* const* d_in, const int* in_sizes, int n_in,
                              void* d_out, int out_size, void* d_ws,
                              size_t ws_size, hipStream_t stream) {
  const float* query = (const float*)d_in[0];
  const float* key   = (const float*)d_in[1];
  const float* value = (const float*)d_in[2];
  const int*   mask  = (const int*)d_in[3];
  const float* WQ    = (const float*)d_in[4];
  const float* WK    = (const float*)d_in[5];
  const float* WV    = (const float*)d_in[6];
  const float* WO    = (const float*)d_in[7];
  const float* bO    = (const float*)d_in[8];
  const float* gamma = (const float*)d_in[9];
  const float* beta  = (const float*)d_in[10];
  float* out = (float*)d_out;

  size_t per = (size_t)B_SZ * N_SZ * D_SZ;  // 4,194,304 elements
  size_t wsz = (size_t)D_SZ * D_SZ;         // 262,144 elements
  unsigned short* p   = (unsigned short*)d_ws;
  unsigned short* qbf = p; p += per;
  unsigned short* kbf = p; p += per;
  unsigned short* vbf = p; p += per;
  unsigned short* WQb = p; p += wsz;
  unsigned short* WKb = p; p += wsz;
  unsigned short* WVb = p; p += wsz;
  unsigned short* WOb = p; p += wsz;
  unsigned short* Qw  = p; p += per;
  unsigned short* Kw  = p; p += per;
  unsigned short* Vt  = p; p += per;
  unsigned short* AO  = p; p += per;   // total ~58 MiB

  // bulk fp32->bf16 conversions (each size divisible by 512)
  f32_to_bf16_kernel<<<dim3((unsigned)(per / 512)), 256, 0, stream>>>(query, qbf, (int)per);
  f32_to_bf16_kernel<<<dim3((unsigned)(per / 512)), 256, 0, stream>>>(key,   kbf, (int)per);
  f32_to_bf16_kernel<<<dim3((unsigned)(per / 512)), 256, 0, stream>>>(value, vbf, (int)per);
  f32_to_bf16_kernel<<<dim3((unsigned)(wsz / 512)), 256, 0, stream>>>(WQ, WQb, (int)wsz);
  f32_to_bf16_kernel<<<dim3((unsigned)(wsz / 512)), 256, 0, stream>>>(WK, WKb, (int)wsz);
  f32_to_bf16_kernel<<<dim3((unsigned)(wsz / 512)), 256, 0, stream>>>(WV, WVb, (int)wsz);
  f32_to_bf16_kernel<<<dim3((unsigned)(wsz / 512)), 256, 0, stream>>>(WO, WOb, (int)wsz);

  qkv_proj_kernel<<<dim3((B_SZ * N_SZ) / 16, 3), 256, 0, stream>>>(
      qbf, kbf, vbf, WQb, WKb, WVb, Qw, Kw, Vt);
  attn_kernel<<<dim3(N_SZ / 64, H_SZ, B_SZ), 128, 0, stream>>>(
      Qw, Kw, Vt, mask, AO);
  oproj_ln_kernel<<<dim3((B_SZ * N_SZ) / 16), 256, 0, stream>>>(
      AO, WOb, bO, query, gamma, beta, out);
}
